// SlotAttention_65025804862057
// MI455X (gfx1250) — matrix operand order repaired and checked by reading, not compiled, not run on hardware
//
#include <hip/hip_runtime.h>
#include <hip/hip_bf16.h>
#include <math.h>
#include <stdint.h>

// ---------------- problem constants ----------------
#define B_   8
#define N_   4096
#define D_   512
#define H_   128
#define S_   256          // 2*H slots
#define TOPK1 64
#define K2    85          // S // 3
#define SCALE_ 0.08838834764831845f   // H^-0.5
#define LN_EPS_ 1e-5f

typedef __bf16 bf16;
typedef __attribute__((ext_vector_type(16))) __bf16 v16bf;
typedef __attribute__((ext_vector_type(8)))  float  v8f;

// ---------------- CDNA5 async global->LDS helpers ----------------
__device__ __forceinline__ void async_ld_b128(const void* gaddr, unsigned lds_off) {
  // vdst = VGPR with LDS byte offset; vaddr = 64-bit global address (ISA 15.18.3 op 98)
  asm volatile("global_load_async_to_lds_b128 %0, %1, off"
               :: "v"(lds_off), "v"(gaddr) : "memory");
}
__device__ __forceinline__ void wait_async0() {
  asm volatile("s_wait_asynccnt 0x0" ::: "memory");
}

// ---------------- GEMM (bf16 WMMA, f32 accumulate) ----------------
// C[M,N] = A[M,K] * Bt[N,K]^T. Block tile 128x64, K-step 32, double-buffered
// async-LDS staging. 256 threads = 8 waves; wave (wm 0..3, wn 0..1) owns a
// 32x32 tile as 2x2 v_wmma_f32_16x16x32_bf16 accumulators.
#define BLK_M 128
#define BLK_N 64
#define BLK_K 32
#define A_LDS 40   // padded LDS row stride (bf16 elems): 80B -> conflict-free b128 reads
#define B_LDSK 40

// EPI: 0 = none, 1 = +bias, 2 = +bias then exact GELU.  OB/OF: write bf16/f32.
template<int EPI, bool OB, bool OF>
__global__ __launch_bounds__(256) void gemm_bt_wmma(
    const bf16* __restrict__ A, long sA, int lda,
    const bf16* __restrict__ Bt, long sB, int ldb,
    const float* __restrict__ bias,
    bf16* __restrict__ Cb, float* __restrict__ Cf, long sC, int ldc,
    int Kdim, float alpha)
{
  __shared__ bf16 As[2][BLK_M * A_LDS];
  __shared__ bf16 Bs[2][BLK_N * B_LDSK];   // stored [n][k] so frags are contiguous

  const int bz = blockIdx.z;
  A  += (long)bz * sA;
  Bt += (long)bz * sB;
  const int m0 = blockIdx.y * BLK_M;
  const int n0 = blockIdx.x * BLK_N;
  const int tid  = threadIdx.x;
  const int lane = tid & 31;
  const int wave = tid >> 5;
  const int wm = wave & 3;     // 32-row group
  const int wn = wave >> 2;    // 32-col group

  v8f acc[2][2];
#pragma unroll
  for (int i = 0; i < 2; i++)
#pragma unroll
    for (int j = 0; j < 2; j++)
#pragma unroll
      for (int r = 0; r < 8; r++) acc[i][j][r] = 0.f;

  // staging assignments: A: 32B per thread (2 async b128); B: 16B (1 async b128)
  const int ar = tid >> 1, ah = tid & 1;   // row 0..127, 16-elem half
  const int nn = tid >> 2, kc = tid & 3;   // Bt row 0..63, 8-elem k chunk

  auto stage = [&](int kk, int buf) {
    const bf16* ga = A + (long)(m0 + ar) * lda + kk + ah * 16;
    unsigned la = (unsigned)(uintptr_t)&As[buf][ar * A_LDS + ah * 16];
    async_ld_b128(ga, la);
    async_ld_b128(ga + 8, la + 16);
    const bf16* gb = Bt + (long)(n0 + nn) * ldb + kk + kc * 8;
    unsigned lb = (unsigned)(uintptr_t)&Bs[buf][nn * B_LDSK + kc * 8];
    async_ld_b128(gb, lb);
  };

  stage(0, 0);
  int p = 0;
  for (int k0 = 0; k0 < Kdim; k0 += BLK_K) {
    wait_async0();        // this wave's tile-p data landed in LDS
    __syncthreads();      // everyone's landed; everyone done reading buf p^1
    if (k0 + BLK_K < Kdim) stage(k0 + BLK_K, p ^ 1);   // DMA next tile during compute

    const int row16 = lane & 15;
    const int hsel  = lane >> 4;
#pragma unroll
    for (int mt = 0; mt < 2; mt++) {
      // A frag (ISA 7.12.2): lanes 0-15 K 0..7 & 16..23, lanes 16-31 K 8..15 & 24..31
      union { v16bf v; uint4 q[2]; } af;
      const bf16* ap = &As[p][(wm * 32 + mt * 16 + row16) * A_LDS + hsel * 8];
      af.q[0] = *(const uint4*)(ap);
      af.q[1] = *(const uint4*)(ap + 16);
#pragma unroll
      for (int nt = 0; nt < 2; nt++) {
        // B frag: lane n = col; lanes 0-15 K 0..7 & 16..23, lanes 16-31 K 8..15 & 24..31
        union { v16bf v; uint4 q[2]; } bfr;
        const bf16* bp = &Bs[p][(wn * 32 + nt * 16 + row16) * B_LDSK + hsel * 8];
        bfr.q[0] = *(const uint4*)(bp);
        bfr.q[1] = *(const uint4*)(bp + 16);
        acc[mt][nt] = __builtin_amdgcn_wmma_f32_16x16x32_bf16(
            false, af.v, false, bfr.v, (short)0, acc[mt][nt], false, false);
      }
    }
    p ^= 1;
  }

  // ---- epilogue: alpha, bias, gelu, dual-precision store ----
  const int coln16 = lane & 15;
  const int hsel2  = lane >> 4;
#pragma unroll
  for (int mt = 0; mt < 2; mt++) {
#pragma unroll
    for (int nt = 0; nt < 2; nt++) {
      const int gn  = n0 + wn * 32 + nt * 16 + coln16;
      const int gmb = m0 + wm * 32 + mt * 16 + hsel2 * 8;
      float bvv = 0.f;
      if (EPI >= 1) bvv = bias[gn];
#pragma unroll
      for (int r = 0; r < 8; r++) {
        float v = acc[mt][nt][r] * alpha + bvv;
        if (EPI == 2) v = 0.5f * v * (1.f + erff(v * 0.7071067811865476f));
        const long offc = (long)bz * sC + (long)(gmb + r) * ldc + gn;
        if (OF) Cf[offc] = v;
        if (OB) Cb[offc] = (bf16)v;
      }
    }
  }
}

// ---------------- helpers ----------------
__device__ __forceinline__ unsigned f2key(float f) {
  unsigned u = __float_as_uint(f);
  return (u & 0x80000000u) ? ~u : (u | 0x80000000u);
}

__global__ void f32_to_bf16_k(const float* __restrict__ in, bf16* __restrict__ outp, long n) {
  for (long i = (long)blockIdx.x * blockDim.x + threadIdx.x; i < n;
       i += (long)gridDim.x * blockDim.x)
    outp[i] = (bf16)in[i];
}

// in: [K][Nc] f32 row-major -> out: [Nc][K] bf16 (weight transpose-convert)
__global__ void f32_to_bf16_T_k(const float* __restrict__ in, bf16* __restrict__ outp,
                                int K, int Nc) {
  long i = (long)blockIdx.x * 256 + threadIdx.x;
  if (i >= (long)K * Nc) return;
  int k = (int)(i / Nc), n = (int)(i % Nc);
  outp[(long)n * K + k] = (bf16)in[i];
}

// in: [b][R][C] bf16 -> out: [b][C][R] bf16
__global__ void bf16_transpose_b(const bf16* __restrict__ in, bf16* __restrict__ outp,
                                 int R, int C) {
  const int bz = blockIdx.y;
  long i = (long)blockIdx.x * 256 + threadIdx.x;
  if (i >= (long)R * C) return;
  int r = (int)(i / C), c = (int)(i % C);
  outp[(long)bz * R * C + (long)c * R + r] = in[(long)bz * R * C + i];
}

__global__ void init_slots_k(float* __restrict__ sf, bf16* __restrict__ sb) {
  long i = (long)blockIdx.x * 256 + threadIdx.x;
  const long total = (long)B_ * S_ * H_;
  if (i >= total) return;
  int h = (int)(i % H_);
  int s = (int)((i / H_) % S_);
  float v = (s < H_) ? ((s == h) ? 1.f : 0.f) : (((s - H_) == h) ? -1.f : 0.f);
  sf[i] = v;
  sb[i] = (bf16)v;
}

// ---------------- slot update: top-64 radix select + softmax + sparse attn@V + l2norm ----
__global__ __launch_bounds__(256) void slot_topk_update(
    const float* __restrict__ dots, const float* __restrict__ Vf,
    float* __restrict__ slots_f, bf16* __restrict__ slots_b,
    const float* __restrict__ temp_p, int Nrow)
{
  const int s = blockIdx.x, b = blockIdx.y;
  const int tid = threadIdx.x;
  const float* row = dots + ((long)b * S_ + s) * Nrow;

  __shared__ unsigned hist[256];
  __shared__ unsigned sh_prefix, sh_rem, sel_cnt;
  __shared__ int   sel_idx[TOPK1];
  __shared__ float sel_val[TOPK1];
  __shared__ float red[256];

  if (tid == 0) { sh_prefix = 0u; sh_rem = TOPK1; sel_cnt = 0u; }
  __syncthreads();

  for (int pass = 3; pass >= 0; pass--) {
    hist[tid] = 0u;
    __syncthreads();
    const unsigned prefix = sh_prefix;
    const unsigned shift = pass * 8;
    const unsigned pmask = (pass == 3) ? 0u : (0xFFFFFFFFu << (shift + 8));
    for (int i = tid; i < Nrow; i += 256) {
      unsigned k = f2key(row[i]);
      if ((k & pmask) == prefix) atomicAdd(&hist[(k >> shift) & 255u], 1u);
    }
    __syncthreads();
    if (tid == 0) {
      unsigned rem = sh_rem, acc = 0u;
      int bin = 255;
      for (; bin >= 0; bin--) {
        unsigned c = hist[bin];
        if (acc + c >= rem) break;
        acc += c;
      }
      if (bin < 0) bin = 0;
      sh_prefix = prefix | ((unsigned)bin << shift);
      sh_rem = rem - acc;
    }
    __syncthreads();
  }
  const unsigned T = sh_prefix;

  for (int i = tid; i < Nrow; i += 256) {
    unsigned k = f2key(row[i]);
    if (k > T) {
      unsigned pp = atomicAdd(&sel_cnt, 1u);
      if (pp < TOPK1) { sel_idx[pp] = i; sel_val[pp] = row[i]; }
    }
  }
  __syncthreads();
  for (int i = tid; i < Nrow; i += 256) {
    unsigned k = f2key(row[i]);
    if (k == T) {
      unsigned pp = atomicAdd(&sel_cnt, 1u);
      if (pp < TOPK1) { sel_idx[pp] = i; sel_val[pp] = row[i]; }
    }
  }
  __syncthreads();

  const float t1 = temp_p[0];
  red[tid] = (tid < TOPK1) ? sel_val[tid] : -3.0e38f;
  __syncthreads();
  for (int st = 128; st >= 1; st >>= 1) {
    if (tid < st) red[tid] = fmaxf(red[tid], red[tid + st]);
    __syncthreads();
  }
  const float vmax = red[0];
  __syncthreads();
  const float e = (tid < TOPK1) ? expf((sel_val[tid] - vmax) / t1) : 0.f;
  red[tid] = e;
  __syncthreads();
  for (int st = 128; st >= 1; st >>= 1) {
    if (tid < st) red[tid] += red[tid + st];
    __syncthreads();
  }
  const float esum = red[0];
  __syncthreads();
  if (tid < TOPK1) sel_val[tid] = e / esum;
  __syncthreads();

  float outv = 0.f;
  if (tid < H_) {
#pragma unroll 4
    for (int i = 0; i < TOPK1; i++)
      outv += sel_val[i] * Vf[((long)b * N_ + sel_idx[i]) * H_ + tid];
  }
  red[tid] = (tid < H_) ? outv * outv : 0.f;
  __syncthreads();
  for (int st = 128; st >= 1; st >>= 1) {
    if (tid < st) red[tid] += red[tid + st];
    __syncthreads();
  }
  const float inv = 1.f / fmaxf(sqrtf(red[0]), 1e-12f);
  if (tid < H_) {
    const float y = outv * inv;
    const long o = ((long)b * S_ + s) * H_ + tid;
    slots_f[o] = y;
    slots_b[o] = (bf16)y;
  }
}

// ---------------- final attention: top-85-of-256 softmax, dense bf16 scatter ----
__global__ __launch_bounds__(256) void final_topk_scatter(
    const float* __restrict__ logits, bf16* __restrict__ atten,
    const float* __restrict__ temp_p)
{
  const int n = blockIdx.x, b = blockIdx.y;
  const int tid = threadIdx.x;
  const float v = logits[((long)b * N_ + n) * S_ + tid];
  const unsigned key = f2key(v);

  __shared__ unsigned hist[256];
  __shared__ unsigned sh_prefix, sh_rem, tie_cnt;
  __shared__ float red[256];
  if (tid == 0) { sh_prefix = 0u; sh_rem = K2; tie_cnt = 0u; }
  __syncthreads();

  for (int pass = 3; pass >= 0; pass--) {
    hist[tid] = 0u;
    __syncthreads();
    const unsigned shift = pass * 8;
    const unsigned pmask = (pass == 3) ? 0u : (0xFFFFFFFFu << (shift + 8));
    if ((key & pmask) == sh_prefix) atomicAdd(&hist[(key >> shift) & 255u], 1u);
    __syncthreads();
    if (tid == 0) {
      unsigned rem = sh_rem, acc = 0u;
      int bin = 255;
      for (; bin >= 0; bin--) {
        unsigned c = hist[bin];
        if (acc + c >= rem) break;
        acc += c;
      }
      if (bin < 0) bin = 0;
      sh_prefix = sh_prefix | ((unsigned)bin << shift);
      sh_rem = rem - acc;
    }
    __syncthreads();
  }
  const unsigned T = sh_prefix;
  bool sel = (key > T);
  if (key == T) {
    unsigned pp = atomicAdd(&tie_cnt, 1u);
    if (pp < sh_rem) sel = true;
  }
  __syncthreads();

  const float t2 = temp_p[0];
  red[tid] = sel ? v : -3.0e38f;
  __syncthreads();
  for (int st = 128; st >= 1; st >>= 1) {
    if (tid < st) red[tid] = fmaxf(red[tid], red[tid + st]);
    __syncthreads();
  }
  const float vmax = red[0];
  __syncthreads();
  const float e = sel ? expf((v - vmax) / t2) : 0.f;
  red[tid] = e;
  __syncthreads();
  for (int st = 128; st >= 1; st >>= 1) {
    if (tid < st) red[tid] += red[tid + st];
    __syncthreads();
  }
  atten[((long)b * N_ + n) * S_ + tid] = (bf16)(e / red[0]);
}

// ---------------- fused residual-add + layernorm (D = 512) ----------------
__global__ __launch_bounds__(256) void add_ln512(
    const float* __restrict__ x1, const float* __restrict__ x2,
    const float* __restrict__ g, const float* __restrict__ be,
    float* __restrict__ of, bf16* __restrict__ ob)
{
  const long row = blockIdx.x;
  const int tid = threadIdx.x;
  __shared__ float red[256];
  const float* p1 = x1 + row * D_;
  const float* p2 = x2 + row * D_;
  const float xa = p1[tid] + p2[tid];
  const float xb = p1[tid + 256] + p2[tid + 256];
  red[tid] = xa + xb;
  __syncthreads();
  for (int st = 128; st >= 1; st >>= 1) {
    if (tid < st) red[tid] += red[tid + st];
    __syncthreads();
  }
  const float mean = red[0] * (1.f / D_);
  __syncthreads();
  const float da = xa - mean, db = xb - mean;
  red[tid] = da * da + db * db;
  __syncthreads();
  for (int st = 128; st >= 1; st >>= 1) {
    if (tid < st) red[tid] += red[tid + st];
    __syncthreads();
  }
  const float inv = rsqrtf(red[0] * (1.f / D_) + LN_EPS_);
  const float ya = da * inv * g[tid] + be[tid];
  const float yb = db * inv * g[tid + 256] + be[tid + 256];
  if (of) { of[row * D_ + tid] = ya; of[row * D_ + tid + 256] = yb; }
  if (ob) { ob[row * D_ + tid] = (bf16)ya; ob[row * D_ + tid + 256] = (bf16)yb; }
}

// ---------------- host orchestration ----------------
extern "C" void kernel_launch(void* const* d_in, const int* in_sizes, int n_in,
                              void* d_out, int out_size, void* d_ws, size_t ws_size,
                              hipStream_t stream)
{
  (void)in_sizes; (void)n_in; (void)out_size; (void)ws_size;
  const float* X     = (const float*)d_in[0];
  const float* Wk    = (const float*)d_in[1];
  const float* bk    = (const float*)d_in[2];
  const float* Wv    = (const float*)d_in[3];
  const float* bv    = (const float*)d_in[4];
  const float* Wq    = (const float*)d_in[5];
  const float* bq    = (const float*)d_in[6];
  const float* Wback = (const float*)d_in[7];
  const float* bback = (const float*)d_in[8];
  const float* W1    = (const float*)d_in[9];
  const float* b1    = (const float*)d_in[10];
  const float* W2    = (const float*)d_in[11];
  const float* b2    = (const float*)d_in[12];
  const float* ln1g  = (const float*)d_in[13];
  const float* ln1b  = (const float*)d_in[14];
  const float* ln2g  = (const float*)d_in[15];
  const float* ln2b  = (const float*)d_in[16];
  const float* temp1 = (const float*)d_in[17];
  const float* temp2 = (const float*)d_in[18];
  float* outp = (float*)d_out;

  char* base = (char*)d_ws;
  size_t off = 0;
  auto alloc = [&](size_t bytes) -> void* {
    void* p = base + off;
    off += (bytes + 255) & ~(size_t)255;
    return p;
  };

  const long BN = (long)B_ * N_;
  bf16*  Xb      = (bf16*) alloc(BN * D_ * 2);
  bf16*  WkT     = (bf16*) alloc((size_t)H_ * D_ * 2);      // [H][D]
  bf16*  WvT     = (bf16*) alloc((size_t)H_ * D_ * 2);
  bf16*  WqT     = (bf16*) alloc((size_t)H_ * D_ * 2);
  bf16*  WbackT  = (bf16*) alloc((size_t)D_ * H_ * 2);      // [D][H]
  bf16*  W1T     = (bf16*) alloc((size_t)2 * D_ * D_ * 2);  // [2D][D]
  bf16*  W2T     = (bf16*) alloc((size_t)D_ * 2 * D_ * 2);  // [D][2D]
  bf16*  Kb      = (bf16*) alloc(BN * H_ * 2);              // [BN][H] (Bt for dots)
  bf16*  Qeb     = (bf16*) alloc(BN * H_ * 2);
  float* Vf      = (float*)alloc(BN * H_ * 4);
  float* dots    = (float*)alloc((size_t)B_ * S_ * N_ * 4);
  float* slots_f = (float*)alloc((size_t)B_ * S_ * H_ * 4);
  bf16*  slots_b = (bf16*) alloc((size_t)B_ * S_ * H_ * 2); // [S][H] (Bt for logits)
  bf16*  slotsVb = (bf16*) alloc((size_t)B_ * S_ * D_ * 2); // [b][S][D]
  bf16*  slotsVT = (bf16*) alloc((size_t)B_ * D_ * S_ * 2); // [b][D][S] (Bt)
  float* logitsb = (float*)alloc((size_t)B_ * N_ * S_ * 4);
  bf16*  attenb  = (bf16*) alloc((size_t)B_ * N_ * S_ * 2);
  float* embref  = (float*)alloc(BN * D_ * 4);
  float* emb_f   = (float*)alloc(BN * D_ * 4);
  bf16*  emb_b   = (bf16*) alloc(BN * D_ * 2);
  bf16*  h1b     = (bf16*) alloc(BN * 2 * D_ * 2);
  float* h2f     = (float*)alloc(BN * D_ * 4);

  const dim3 blk(256);

  // conversions (+ weight transposes so all GEMMs use the Bt fast path)
  f32_to_bf16_k<<<2048, 256, 0, stream>>>(X, Xb, BN * D_);
  f32_to_bf16_T_k<<<(D_ * H_) / 256, 256, 0, stream>>>(Wk, WkT, D_, H_);
  f32_to_bf16_T_k<<<(D_ * H_) / 256, 256, 0, stream>>>(Wv, WvT, D_, H_);
  f32_to_bf16_T_k<<<(D_ * H_) / 256, 256, 0, stream>>>(Wq, WqT, D_, H_);
  f32_to_bf16_T_k<<<(H_ * D_) / 256, 256, 0, stream>>>(Wback, WbackT, H_, D_);
  f32_to_bf16_T_k<<<(D_ * 2 * D_) / 256, 256, 0, stream>>>(W1, W1T, D_, 2 * D_);
  f32_to_bf16_T_k<<<(2 * D_ * D_) / 256, 256, 0, stream>>>(W2, W2T, 2 * D_, D_);

  // Projections: [BN,512] @ [512,128] + bias
  const dim3 gP(H_ / BLK_N, (unsigned)(BN / BLK_M), 1);
  gemm_bt_wmma<1, true, false><<<gP, blk, 0, stream>>>(
      Xb, 0, D_, WkT, 0, D_, bk, Kb, nullptr, 0, H_, D_, 1.f);
  gemm_bt_wmma<1, false, true><<<gP, blk, 0, stream>>>(
      Xb, 0, D_, WvT, 0, D_, bv, nullptr, Vf, 0, H_, D_, 1.f);
  gemm_bt_wmma<1, true, false><<<gP, blk, 0, stream>>>(
      Xb, 0, D_, WqT, 0, D_, bq, Qeb, nullptr, 0, H_, D_, 1.f);

  init_slots_k<<<(B_ * S_ * H_) / 256, 256, 0, stream>>>(slots_f, slots_b);

  for (int it = 0; it < 2; it++) {
    const dim3 gD(N_ / BLK_N, S_ / BLK_M, B_);   // dots = slots @ K^T * SCALE
    gemm_bt_wmma<0, false, true><<<gD, blk, 0, stream>>>(
        slots_b, (long)S_ * H_, H_, Kb, (long)N_ * H_, H_, nullptr,
        nullptr, dots, (long)S_ * N_, N_, H_, SCALE_);
    slot_topk_update<<<dim3(S_, B_), 256, 0, stream>>>(
        dots, Vf, slots_f, slots_b, temp1, N_);
  }

  // slots_V = slots @ Wback + bback (flattened over batch), then per-batch transpose
  const dim3 gSV(D_ / BLK_N, (B_ * S_) / BLK_M, 1);
  gemm_bt_wmma<1, true, false><<<gSV, blk, 0, stream>>>(
      slots_b, 0, H_, WbackT, 0, H_, bback, slotsVb, nullptr, 0, D_, H_, 1.f);
  bf16_transpose_b<<<dim3((S_ * D_) / 256, B_), 256, 0, stream>>>(
      slotsVb, slotsVT, S_, D_);

  // logits = embQ @ slots^T * SCALE
  const dim3 gL(S_ / BLK_N, N_ / BLK_M, B_);
  gemm_bt_wmma<0, false, true><<<gL, blk, 0, stream>>>(
      Qeb, (long)N_ * H_, H_, slots_b, (long)S_ * H_, H_, nullptr,
      nullptr, logitsb, (long)N_ * S_, S_, H_, SCALE_);

  final_topk_scatter<<<dim3(N_, B_), 256, 0, stream>>>(logitsb, attenb, temp2);

  // emb_refined = atten @ slots_V   (Bt = slotsVT [D][S])
  const dim3 gE(D_ / BLK_N, N_ / BLK_M, B_);
  gemm_bt_wmma<0, false, true><<<gE, blk, 0, stream>>>(
      attenb, (long)N_ * S_, S_, slotsVT, (long)D_ * S_, S_, nullptr,
      nullptr, embref, (long)N_ * D_, D_, S_, 1.f);

  // emb = LN1(emb_refined + inputs)
  add_ln512<<<(unsigned)BN, 256, 0, stream>>>(embref, X, ln1g, ln1b, emb_f, emb_b);

  // FFN: gelu(emb @ W1 + b1) @ W2 + b2
  const dim3 g1((2 * D_) / BLK_N, (unsigned)(BN / BLK_M), 1);
  gemm_bt_wmma<2, true, false><<<g1, blk, 0, stream>>>(
      emb_b, 0, D_, W1T, 0, D_, b1, h1b, nullptr, 0, 2 * D_, D_, 1.f);
  const dim3 g2(D_ / BLK_N, (unsigned)(BN / BLK_M), 1);
  gemm_bt_wmma<1, false, true><<<g2, blk, 0, stream>>>(
      h1b, 0, 2 * D_, W2T, 0, 2 * D_, b2, nullptr, h2f, 0, D_, 2 * D_, 1.f);

  // out = LN2(emb + ffn)
  add_ln512<<<(unsigned)BN, 256, 0, stream>>>(emb_f, h2f, ln2g, ln2b, outp, nullptr);
}
